// ExpertChoiceFF_13477607375856
// MI455X (gfx1250) — matrix operand order repaired and checked by reading, not compiled, not run on hardware
//
#include <hip/hip_runtime.h>
#include <hip/hip_bf16.h>

#define BB   8
#define CC   4096
#define DD   1024
#define EE   16
#define ESS  256
#define KSEL 4096
#define NTOK 32768

typedef __attribute__((ext_vector_type(16))) __bf16 v16bf;
typedef __attribute__((ext_vector_type(8)))  float  v8f;

union Frag { v16bf v; unsigned u[8]; };

__device__ __forceinline__ unsigned short f2bf(float f) {
    unsigned u = __float_as_uint(f);
    u += 0x7FFFu + ((u >> 16) & 1u);          // round-to-nearest-even
    return (unsigned short)(u >> 16);
}
__device__ __forceinline__ unsigned packbf2(float a, float b) {
    return (unsigned)f2bf(a) | ((unsigned)f2bf(b) << 16);
}
__device__ __forceinline__ unsigned orderKey(float f) {
    unsigned u = __float_as_uint(f);
    return u ^ ((u & 0x80000000u) ? 0xFFFFFFFFu : 0x80000000u);
}
// 16-bit A-matrix K map (ISA 7.12.2): VGPR v holds K pairs, lane group g adds +8
__device__ __forceinline__ int kmapA(int v, int g) {
    return ((v < 4) ? 2 * v : 16 + 2 * (v - 4)) + 8 * g;
}

// CDNA5 async global->LDS copy, 16 bytes per lane, tracked by ASYNCcnt.
// GV addressing: 64-bit global address in VGPR pair, LDS byte address in VGPR.
__device__ __forceinline__ void async_copy16(unsigned lds_addr, const void* gptr) {
    asm volatile("global_load_async_to_lds_b128 %0, %1, off"
                 :: "v"(lds_addr), "v"((unsigned long long)gptr)
                 : "memory");
}
__device__ __forceinline__ void wait_async0() {
    asm volatile("s_wait_asynccnt 0x0" ::: "memory");
}
__device__ __forceinline__ unsigned lds_off(const void* p) {
    return (unsigned)(unsigned long long)p;   // generic LDS addr: low 32 bits = LDS offset
}

// ---------------- kernel 0: zero output ----------------
__global__ __launch_bounds__(1024) void zero_kernel(float4* z) {
    long i = (long)blockIdx.x * 1024 + threadIdx.x;
    z[i] = make_float4(0.f, 0.f, 0.f, 0.f);
}

// ---------------- kernel 1: row-sum gate logits ----------------
// logits[e][n] = (sum_d x[n,d]) * gate[n % C, e]
__global__ __launch_bounds__(256) void gate_kernel(const float* __restrict__ x,
                                                   const float* __restrict__ gate,
                                                   float* __restrict__ logits) {
    int t = threadIdx.x, wv = t >> 5, lane = t & 31;
    int token = blockIdx.x * 8 + wv;
    const float4* xr = reinterpret_cast<const float4*>(x + (long)token * DD);
    float s = 0.f;
    for (int i = 0; i < 8; ++i) {
        float4 v = xr[i * 32 + lane];
        s += v.x + v.y + v.z + v.w;
    }
    for (int off = 16; off > 0; off >>= 1) s += __shfl_xor(s, off, 32);
    if (lane < EE) {
        int c = token & (CC - 1);
        logits[lane * NTOK + token] = s * gate[c * EE + lane];
    }
}

// ---------------- kernel 2: per-expert softmax stats ----------------
__global__ __launch_bounds__(1024) void stats_kernel(const float* __restrict__ logits,
                                                     float* __restrict__ mx,
                                                     float* __restrict__ sm) {
    __shared__ float red[1024];
    int e = blockIdx.x, t = threadIdx.x;
    float m = -3.4e38f;
    for (int i = 0; i < 32; ++i) m = fmaxf(m, logits[e * NTOK + i * 1024 + t]);
    red[t] = m; __syncthreads();
    for (int s = 512; s > 0; s >>= 1) { if (t < s) red[t] = fmaxf(red[t], red[t + s]); __syncthreads(); }
    float mval = red[0]; __syncthreads();
    float su = 0.f;
    for (int i = 0; i < 32; ++i) su += __expf(logits[e * NTOK + i * 1024 + t] - mval);
    red[t] = su; __syncthreads();
    for (int s = 512; s > 0; s >>= 1) { if (t < s) red[t] += red[t + s]; __syncthreads(); }
    if (t == 0) { mx[e] = mval; sm[e] = red[0]; }
}

// ---------------- kernel 3: radix-histogram threshold for top-K ----------------
__global__ __launch_bounds__(1024) void select_kernel(const float* __restrict__ logits,
                                                      unsigned* __restrict__ thresh,
                                                      unsigned* __restrict__ cntHi,
                                                      unsigned* __restrict__ cntEq) {
    __shared__ unsigned hist[4096];
    int e = blockIdx.x, t = threadIdx.x;
    for (int i = 0; i < 4; ++i) hist[i * 1024 + t] = 0u;
    __syncthreads();
    for (int i = 0; i < 32; ++i) {
        unsigned k = orderKey(logits[e * NTOK + i * 1024 + t]);
        atomicAdd(&hist[k >> 20], 1u);
    }
    __syncthreads();
    if (t == 0) {
        unsigned cum = 0; int b = 4095;
        for (; b >= 0; --b) {
            unsigned c = hist[b];
            if (cum + c >= (unsigned)KSEL) break;
            cum += c;
        }
        thresh[e] = (unsigned)b;
        cntHi[e] = 0u;        // slots [0, cum) for bins > threshold
        cntEq[e] = cum;       // slots [cum, K) for threshold bin
    }
}

// ---------------- kernel 4: compaction ----------------
__global__ __launch_bounds__(256) void compact_kernel(const float* __restrict__ logits,
                                                      const float* __restrict__ mx,
                                                      const float* __restrict__ sm,
                                                      const unsigned* __restrict__ thresh,
                                                      unsigned* __restrict__ cntHi,
                                                      unsigned* __restrict__ cntEq,
                                                      int* __restrict__ tki,
                                                      float* __restrict__ tkp) {
    int e = blockIdx.y;
    int tok = blockIdx.x * 256 + threadIdx.x;
    float l = logits[e * NTOK + tok];
    unsigned key = orderKey(l);
    unsigned bin = key >> 20;
    unsigned T = thresh[e];
    int slot = -1;
    if (bin > T) {
        slot = (int)atomicAdd(&cntHi[e], 1u);
    } else if (bin == T) {
        unsigned q = atomicAdd(&cntEq[e], 1u);
        if (q < (unsigned)KSEL) slot = (int)q;
    }
    if (slot >= 0) {
        tki[e * KSEL + slot] = tok;
        tkp[e * KSEL + slot] = __expf(l - mx[e]) / sm[e];
    }
}

// ---------------- kernel 5a/5b: one-shot weight convert + WMMA-fragment swizzle ----------------
// w1s layout: [e][kc(16)][ks(2)][nt(16)][lane(32)][8 dwords]   (8192 dwords per (e,kc))
__global__ __launch_bounds__(256) void wswz1_kernel(const float* __restrict__ w1,
                                                    unsigned* __restrict__ w1s) {
    int kc = blockIdx.x, e = blockIdx.y, t = threadIdx.x;
    for (int i = 0; i < 32; ++i) {
        int wl = t * 32 + i;
        int v  = wl & 7;
        int sl = (wl >> 3) & 31;
        int nt = (wl >> 8) & 15;
        int ks = (wl >> 12) & 1;
        int g  = sl >> 4;
        int s  = nt * 16 + (sl & 15);
        int d  = kc * 64 + ks * 32 + kmapA(v, g);
        float a0 = w1[(long)d * (EE * ESS) + e * ESS + s];
        float a1 = w1[(long)(d + 1) * (EE * ESS) + e * ESS + s];
        w1s[(long)((e * 16 + kc) << 13) + ((((ks * 16 + nt) * 32 + sl) << 3) + v)] = packbf2(a0, a1);
    }
}
// w2s layout: [e][nc(8)][ks2(8)][nt(8)][lane(32)][8 dwords]    (16384 dwords per (e,nc))
__global__ __launch_bounds__(256) void wswz2_kernel(const float* __restrict__ w2,
                                                    unsigned* __restrict__ w2s) {
    int nc = blockIdx.x, e = blockIdx.y, t = threadIdx.x;
    for (int i = 0; i < 64; ++i) {
        int wl  = t * 64 + i;
        int v   = wl & 7;
        int sl  = (wl >> 3) & 31;
        int nt  = (wl >> 8) & 7;
        int ks2 = (wl >> 11) & 7;
        int g   = sl >> 4;
        int s   = ks2 * 32 + kmapA(v, g);               // ES (K) index
        int dcol = nc * 128 + nt * 16 + (sl & 15);
        float a0 = w2[(long)e * ESS * DD + (long)s * DD + dcol];
        float a1 = w2[(long)e * ESS * DD + (long)(s + 1) * DD + dcol];
        w2s[(long)((e * 8 + nc) << 14) + ((((ks2 * 8 + nt) * 32 + sl) << 3) + v)] = packbf2(a0, a1);
    }
}

// ---------------- kernel 6: fused gather + GEMM1(relu) + GEMM2 + scaled scatter-add ----------------
// block = (expert, 128-token tile); 8 waves; bf16 WMMA 16x16x32 f32-accumulate.
// Weights arrive pre-swizzled; staged into LDS with async global->LDS b128 DMA.
__global__ __launch_bounds__(256) void ffn_kernel(const float* __restrict__ x,
                                                  const unsigned* __restrict__ w1s,
                                                  const unsigned* __restrict__ w2s,
                                                  const int* __restrict__ tki_g,
                                                  const float* __restrict__ tkp_g,
                                                  float* __restrict__ z) {
    __shared__ __align__(16) unsigned short xs[128 * 64];   // 16 KB : gathered x chunk (bf16)
    __shared__ __align__(16) unsigned       wbuf[16384];    // 64 KB : B-fragments (w1 / w2)
    __shared__ __align__(16) unsigned short hs[128 * 256];  // 64 KB : h = relu(x @ w1) (bf16)
    __shared__ int   tki[128];
    __shared__ float pvl[128];

    const int t = threadIdx.x;
    const int e = blockIdx.y;
    const int tile = blockIdx.x;
    const int wv = t >> 5;
    const int lane = t & 31;
    const int lg = lane >> 4;
    const int ln = lane & 15;
    const unsigned wbuf_lds = lds_off(wbuf);

    if (t < 128) {
        tki[t] = tki_g[e * KSEL + tile * 128 + t];
        pvl[t] = tkp_g[e * KSEL + tile * 128 + t];
    }
    __syncthreads();

    v8f acc[16];
    for (int n = 0; n < 16; ++n)
        for (int j = 0; j < 8; ++j) acc[n][j] = 0.f;

    // -------- GEMM1: [128 tok x 1024] @ w1[e] [1024 x 256], K chunks of 64 --------
    for (int kc = 0; kc < 16; ++kc) {
        // async DMA: pre-swizzled w1 chunk (8192 dwords) -> wbuf, overlaps with x gather below
        {
            const char* g = (const char*)(w1s + ((long)(e * 16 + kc) << 13)) + t * 128;
            unsigned l = wbuf_lds + (unsigned)t * 128u;
            for (int i = 0; i < 8; ++i)
                async_copy16(l + i * 16, g + i * 16);
        }
        // stage A: gather 128 rows x 64 cols of x -> bf16 LDS (row-major)
        for (int i = 0; i < 8; ++i) {
            int fl = t * 8 + i;                 // 2048 float4s
            int row = fl >> 4;
            int j = fl & 15;
            const float4 vv = *reinterpret_cast<const float4*>(
                x + (long)tki[row] * DD + kc * 64 + j * 4);
            unsigned* dst = reinterpret_cast<unsigned*>(&xs[row * 64 + j * 4]);
            dst[0] = packbf2(vv.x, vv.y);
            dst[1] = packbf2(vv.z, vv.w);
        }
        wait_async0();
        __syncthreads();
        for (int ks = 0; ks < 2; ++ks) {
            Frag af;
            const int M = wv * 16 + ln;
            for (int v = 0; v < 8; ++v)
                af.u[v] = *reinterpret_cast<const unsigned*>(&xs[M * 64 + ks * 32 + kmapA(v, lg)]);
            for (int n = 0; n < 16; ++n) {
                Frag bm;
                const unsigned* src = &wbuf[((ks * 16 + n) * 32 + lane) << 3];
                for (int v = 0; v < 8; ++v) bm.u[v] = src[v];
                acc[n] = __builtin_amdgcn_wmma_f32_16x16x32_bf16(
                    false, af.v, false, bm.v, (short)0, acc[n], false, false);
            }
        }
        __syncthreads();
    }

    // relu + store h as bf16 (each wave owns its own 16 M rows)
    for (int n = 0; n < 16; ++n)
        for (int r = 0; r < 8; ++r) {
            int M = wv * 16 + lg * 8 + r;
            int s = n * 16 + ln;
            float hv = acc[n][r];
            hs[M * 256 + s] = f2bf(hv > 0.f ? hv : 0.f);
        }

    // -------- GEMM2: h [128 x 256] @ w2[e] [256 x 1024], N chunks of 128 --------
    for (int nc = 0; nc < 8; ++nc) {
        __syncthreads();
        // async DMA: pre-swizzled w2 chunk (16384 dwords) -> wbuf
        {
            const char* g = (const char*)(w2s + ((long)(e * 8 + nc) << 14)) + t * 256;
            unsigned l = wbuf_lds + (unsigned)t * 256u;
            for (int i = 0; i < 16; ++i)
                async_copy16(l + i * 16, g + i * 16);
        }
        wait_async0();
        __syncthreads();
        v8f acc2[8];
        for (int n = 0; n < 8; ++n)
            for (int j = 0; j < 8; ++j) acc2[n][j] = 0.f;
        for (int ks2 = 0; ks2 < 8; ++ks2) {
            Frag af;
            const int M = wv * 16 + ln;
            for (int v = 0; v < 8; ++v)
                af.u[v] = *reinterpret_cast<const unsigned*>(&hs[M * 256 + ks2 * 32 + kmapA(v, lg)]);
            for (int n = 0; n < 8; ++n) {
                Frag bm;
                const unsigned* src = &wbuf[((ks2 * 8 + n) * 32 + lane) << 3];
                for (int v = 0; v < 8; ++v) bm.u[v] = src[v];
                acc2[n] = __builtin_amdgcn_wmma_f32_16x16x32_bf16(
                    false, af.v, false, bm.v, (short)0, acc2[n], false, false);
            }
        }
        // epilogue: scale by gate prob, scatter-add into z
        for (int n = 0; n < 8; ++n)
            for (int r = 0; r < 8; ++r) {
                int M = wv * 16 + lg * 8 + r;
                int col = nc * 128 + n * 16 + ln;
                float val = acc2[n][r] * pvl[M];
                unsafeAtomicAdd(z + (long)tki[M] * DD + col, val);
            }
    }
}

extern "C" void kernel_launch(void* const* d_in, const int* in_sizes, int n_in,
                              void* d_out, int out_size, void* d_ws, size_t ws_size,
                              hipStream_t stream) {
    (void)in_sizes; (void)n_in; (void)out_size; (void)ws_size;
    const float* x    = (const float*)d_in[0];   // [B,C,D]
    const float* gate = (const float*)d_in[1];   // [C,E]
    const float* w1   = (const float*)d_in[2];   // [D,E,ES]
    const float* w2   = (const float*)d_in[3];   // [E,ES,D]
    float* z = (float*)d_out;                    // [B,C,D]

    char* ws = (char*)d_ws;                      // needs ~20 MB scratch
    float*    logits = (float*)ws;                         // 2 MB
    float*    mx     = (float*)(ws + 2097152);
    float*    sm     = mx + 16;
    unsigned* thresh = (unsigned*)(sm + 16);
    unsigned* cntHi  = thresh + 16;
    unsigned* cntEq  = cntHi + 16;
    int*      tki    = (int*)(ws + 2101248);               // 256 KB
    float*    tkp    = (float*)(ws + 2363392);             // 256 KB
    unsigned* w1s    = (unsigned*)(ws + 4194304);          // 8 MB bf16 swizzled w1
    unsigned* w2s    = (unsigned*)(ws + 12582912);         // 8 MB bf16 swizzled w2

    zero_kernel<<<8192, 1024, 0, stream>>>((float4*)z);
    wswz1_kernel<<<dim3(16, EE), 256, 0, stream>>>(w1, w1s);
    wswz2_kernel<<<dim3(8, EE), 256, 0, stream>>>(w2, w2s);
    gate_kernel<<<NTOK / 8, 256, 0, stream>>>(x, gate, logits);
    stats_kernel<<<EE, 1024, 0, stream>>>(logits, mx, sm);
    select_kernel<<<EE, 1024, 0, stream>>>(logits, thresh, cntHi, cntEq);
    compact_kernel<<<dim3(NTOK / 256, EE), 256, 0, stream>>>(logits, mx, sm, thresh,
                                                             cntHi, cntEq, tki, tkp);
    ffn_kernel<<<dim3(KSEL / 128, EE), 256, 0, stream>>>(x, w1s, w2s, tki, tkp, z);
}
